// SAFM_9929964388443
// MI455X (gfx1250) — matrix operand order repaired and checked by reading, not compiled, not run on hardware
//
#include <hip/hip_runtime.h>
#include <hip/hip_bf16.h>
#include <cstdint>
#include <cstddef>

// ---------------------------------------------------------------------------
// CDNA5 (gfx1250) implementation: wave32 + v_wmma_f32_16x16x32_f16 for every
// matmul; async global->LDS staging (ASYNCcnt) for the large GEMMs;
// non-temporal stores for the 524MB output stream.
// ---------------------------------------------------------------------------

typedef __attribute__((ext_vector_type(16))) _Float16 v16h;
typedef __attribute__((ext_vector_type(8)))  _Float16 v8h;
typedef __attribute__((ext_vector_type(8)))  float    v8f;
typedef __attribute__((ext_vector_type(4)))  float    v4f;

static __device__ __forceinline__ v8f wmma_16x16x32(v16h a, v16h b, v8f c) {
  // (neg_a, A, neg_b, B, c_mod, C, reuse_a, reuse_b)
  return __builtin_amdgcn_wmma_f32_16x16x32_f16(false, a, false, b, (short)0, c, false, false);
}

static __device__ __forceinline__ v16h zero_v16h() {
  v16h z;
#pragma unroll
  for (int i = 0; i < 16; ++i) z[i] = (_Float16)0.f;
  return z;
}
static __device__ __forceinline__ v8f zero_v8f() {
  v8f z;
#pragma unroll
  for (int i = 0; i < 8; ++i) z[i] = 0.f;
  return z;
}
static __device__ __forceinline__ float sigmoidf(float x) {
  return 1.0f / (1.0f + __expf(-x));
}

// LDS byte offset of a __shared__ object (generic -> AS(3) -> int strips the
// aperture and yields the raw LDS address the async-DMA instructions want).
static __device__ __forceinline__ unsigned lds_off_of(const void* p) {
  return (unsigned)(uintptr_t)(const __attribute__((address_space(3))) void*)p;
}

// GLOBAL_LOAD_ASYNC_TO_LDS_B128 (ISA 10.x / 15.18.3 op 98): per-lane async
// DMA of 16B from global to LDS, tracked with ASYNCcnt (no VGPR staging).
static __device__ __forceinline__ void async_copy_b128(unsigned lds_addr,
                                                       const void* gaddr) {
  asm volatile("global_load_async_to_lds_b128 %0, %1, off"
               :: "v"(lds_addr), "v"(gaddr)
               : "memory");
}
static __device__ __forceinline__ void wait_asynccnt0() {
  asm volatile("s_wait_asynccnt 0x0" ::: "memory");
}

// ---------------------------------------------------------------------------
// Exact-shape WMMA GEMM with async-LDS staging.  Requires M%32==0, N%256==0.
//   C[M,N] = A[M,128] @ W[N,128]^T + bias
// Block: 256 threads = 8 waves; block tile 32(M) x 256(N); wave tile 16x64.
// Per block: A-tile 32x128 f16 (8KB) and B-tile 256x128 f16 (64KB) are
// contiguous row ranges in memory -> staged with linear async B128 copies,
// s_wait_asynccnt 0 + barrier, then WMMA fragments are fed from LDS:
//   A 16x32 f16: lane=row(lane&15), 16 contiguous K halves k0..k0+15,
//                k0 = kc*32 + (lane>>4)*16
//   B 32x16 f16: lane=col(lane&15), 16 contiguous K halves at
//                kc*32 + (lane>>4)*16  (one 32B LDS read)
//   C 16x16 f32: vgpr r -> row (lane>>4)*8 + r, col lane&15
// NT=true streams C with non-temporal stores (keep the 192MB L2 for weights).
// ---------------------------------------------------------------------------
template <bool NT>
__global__ __launch_bounds__(256) void gemm_nt_k128_lds(
    const _Float16* __restrict__ A, const _Float16* __restrict__ W,
    const float* __restrict__ bias, float* __restrict__ C, int M, int N) {
  __shared__ _Float16 aT[32][128];    // 8 KB
  __shared__ _Float16 bT[256][128];   // 64 KB

  const int tid  = threadIdx.x;
  const int lane = tid & 31;
  const int wave = tid >> 5;
  const int wm   = wave & 1;
  const int wn   = wave >> 1;
  const int lrow = lane & 15;
  const int lhi  = lane >> 4;

  const int mblock = blockIdx.y * 32;
  const int nblock = blockIdx.x * 256;

  // ---- async stage: A tile (2 x B128/thread), B tile (16 x B128/thread) ----
  {
    const char* ag = (const char*)(A + (size_t)mblock * 128);
    const char* bg = (const char*)(W + (size_t)nblock * 128);
    const unsigned al = lds_off_of(&aT[0][0]);
    const unsigned bl = lds_off_of(&bT[0][0]);
#pragma unroll
    for (int j = 0; j < 2; ++j) {
      const unsigned o = (unsigned)(tid + j * 256) * 16u;
      async_copy_b128(al + o, ag + o);
    }
#pragma unroll
    for (int j = 0; j < 16; ++j) {
      const unsigned o = (unsigned)(tid + j * 256) * 16u;
      async_copy_b128(bl + o, bg + o);
    }
  }
  wait_asynccnt0();
  __syncthreads();

  // ---- WMMA from LDS ----
  v8f acc[4];
#pragma unroll
  for (int nt = 0; nt < 4; ++nt) acc[nt] = zero_v8f();

#pragma unroll
  for (int kc = 0; kc < 4; ++kc) {
    const int k0 = kc * 32 + lhi * 16;
    v8h alo = *(const v8h*)(&aT[wm * 16 + lrow][k0]);
    v8h ahi = *(const v8h*)(&aT[wm * 16 + lrow][k0 + 8]);
    v16h a = __builtin_shufflevector(alo, ahi, 0, 1, 2, 3, 4, 5, 6, 7,
                                     8, 9, 10, 11, 12, 13, 14, 15);
#pragma unroll
    for (int nt = 0; nt < 4; ++nt) {
      v16h b = *(const v16h*)(&bT[wn * 64 + nt * 16 + lrow][kc * 32 + lhi * 16]);
      acc[nt] = wmma_16x16x32(a, b, acc[nt]);
    }
  }

  // ---- store (exact shapes: no guards) ----
#pragma unroll
  for (int nt = 0; nt < 4; ++nt) {
    const int n  = nblock + wn * 64 + nt * 16 + lrow;
    const float bv = bias ? bias[n] : 0.f;
#pragma unroll
    for (int r = 0; r < 8; ++r) {
      const int m = mblock + wm * 16 + lhi * 8 + r;
      const float v = acc[nt][r] + bv;
      if (NT) __builtin_nontemporal_store(v, &C[(size_t)m * N + n]);
      else    C[(size_t)m * N + n] = v;
    }
  }
}

// ---------------------------------------------------------------------------
// Guarded generic WMMA GEMM (used for the GCN layers, N=128).
// Block tile 64(M) x 128(N): wm = wave&3, wn = wave>>2; wave tile 16x64.
// ---------------------------------------------------------------------------
__global__ __launch_bounds__(256) void gemm_nt_k128(
    const _Float16* __restrict__ A, const _Float16* __restrict__ W,
    const float* __restrict__ bias, float* __restrict__ C, int M, int N) {
  const int lane = threadIdx.x & 31;
  const int wave = threadIdx.x >> 5;
  const int wm = wave & 3;
  const int wn = wave >> 2;
  const int lrow = lane & 15;
  const int lhi  = lane >> 4;

  const int mbase = blockIdx.y * 64 + wm * 16;
  const int nbase = blockIdx.x * 128 + wn * 64;

  const int  am    = mbase + lrow;
  const bool am_ok = (am < M);
  const _Float16* arow = A + (size_t)(am_ok ? am : 0) * 128;

  v8f acc[4];
#pragma unroll
  for (int nt = 0; nt < 4; ++nt) acc[nt] = zero_v8f();

#pragma unroll
  for (int kc = 0; kc < 4; ++kc) {
    v16h a;
    if (am_ok) {
      const int k0 = kc * 32 + lhi * 16;
      v8h alo = *(const v8h*)(arow + k0);
      v8h ahi = *(const v8h*)(arow + k0 + 8);
      a = __builtin_shufflevector(alo, ahi, 0, 1, 2, 3, 4, 5, 6, 7,
                                  8, 9, 10, 11, 12, 13, 14, 15);
    } else {
      a = zero_v16h();
    }
#pragma unroll
    for (int nt = 0; nt < 4; ++nt) {
      const int n = nbase + nt * 16 + lrow;
      v16h b = (n < N) ? *(const v16h*)(W + (size_t)n * 128 + kc * 32 + lhi * 16)
                       : zero_v16h();
      acc[nt] = wmma_16x16x32(a, b, acc[nt]);
    }
  }

#pragma unroll
  for (int nt = 0; nt < 4; ++nt) {
    const int n = nbase + nt * 16 + lrow;
    if (n >= N) continue;
    const float bv = bias ? bias[n] : 0.f;
#pragma unroll
    for (int r = 0; r < 8; ++r) {
      const int m = mbase + lhi * 8 + r;
      if (m < M) C[(size_t)m * N + n] = acc[nt][r] + bv;
    }
  }
}

// ---------------------------------------------------------------------------
// Sequential LSTM (B=8 padded to M=16, H=128, S=512).  One workgroup, 8 waves.
// W_hh [512,128] f32 is converted once into WMMA B-fragments RESIDENT IN
// VGPRs (16 frags x 8 VGPRs = 128 VGPRs/wave): the 512-step loop touches no
// HBM for weights.  h round-trips via 4KB f16 LDS; gates via 32KB f32 LDS;
// c-state lives in thread registers.  Wave w owns gate columns [64w,64w+64).
// mode 0: encoder (xg=[8,512,512], writes enc_mean[8,128])
// mode 1: decoder (xg=[8,512] time-constant, writes hs f16 [4096,128])
// ---------------------------------------------------------------------------
__global__ __launch_bounds__(256) void lstm_seq(
    const float* __restrict__ xg, const float* __restrict__ Whh,
    float* __restrict__ enc_mean, _Float16* __restrict__ hs_out, int mode) {
  __shared__ _Float16 hbuf[16][128];
  __shared__ float    gbuf[16][512];

  const int tid  = threadIdx.x;
  const int lane = tid & 31;
  const int wave = tid >> 5;
  const int lrow = lane & 15;
  const int lhi  = lane >> 4;
  const int wb   = wave * 64;

  // one-time: load + convert W_hh^T into resident B fragments
  v16h bw[4][4];
#pragma unroll
  for (int nt = 0; nt < 4; ++nt) {
#pragma unroll
    for (int kc = 0; kc < 4; ++kc) {
      const int n  = wb + nt * 16 + lrow;
      const int kb = kc * 32 + lhi * 16;
      const float* wp = Whh + (size_t)n * 128 + kb;
      v4f w0 = *(const v4f*)(wp);
      v4f w1 = *(const v4f*)(wp + 4);
      v4f w2 = *(const v4f*)(wp + 8);
      v4f w3 = *(const v4f*)(wp + 12);
      v16h b;
#pragma unroll
      for (int i = 0; i < 4; ++i) {
        b[i]      = (_Float16)w0[i];
        b[4 + i]  = (_Float16)w1[i];
        b[8 + i]  = (_Float16)w2[i];
        b[12 + i] = (_Float16)w3[i];
      }
      bw[nt][kc] = b;
    }
  }

  for (int i = tid; i < 16 * 128; i += 256) ((_Float16*)hbuf)[i] = (_Float16)0.f;

  const bool active = (tid < 128);  // threads 0..127 own the 8 real batch rows
  const int  gc = tid & 127;
  float cst[8], hsum[8];
#pragma unroll
  for (int r = 0; r < 8; ++r) { cst[r] = 0.f; hsum[r] = 0.f; }

  __syncthreads();

  for (int t = 0; t < 512; ++t) {
    // WMMA phase: g[16,512] = h[16,128] @ W_hh^T
    v8f acc[4];
#pragma unroll
    for (int nt = 0; nt < 4; ++nt) acc[nt] = zero_v8f();
#pragma unroll
    for (int kc = 0; kc < 4; ++kc) {
      const int k0 = kc * 32 + lhi * 16;
      v8h alo = *(const v8h*)(&hbuf[lrow][k0]);
      v8h ahi = *(const v8h*)(&hbuf[lrow][k0 + 8]);
      v16h a = __builtin_shufflevector(alo, ahi, 0, 1, 2, 3, 4, 5, 6, 7,
                                       8, 9, 10, 11, 12, 13, 14, 15);
#pragma unroll
      for (int nt = 0; nt < 4; ++nt)
        acc[nt] = wmma_16x16x32(a, bw[nt][kc], acc[nt]);
    }
#pragma unroll
    for (int nt = 0; nt < 4; ++nt) {
      const int n = wb + nt * 16 + lrow;
#pragma unroll
      for (int r = 0; r < 8; ++r) gbuf[lhi * 8 + r][n] = acc[nt][r];
    }
    __syncthreads();

    // gate phase (threads 0..127, 8 rows each)
    if (active) {
#pragma unroll
      for (int r = 0; r < 8; ++r) {
        const float* xr = (mode == 0) ? (xg + ((size_t)r * 512 + t) * 512)
                                      : (xg + (size_t)r * 512);
        const float gi = sigmoidf(gbuf[r][gc]       + xr[gc]);
        const float gf = sigmoidf(gbuf[r][128 + gc] + xr[128 + gc]);
        const float gg = tanhf   (gbuf[r][256 + gc] + xr[256 + gc]);
        const float go = sigmoidf(gbuf[r][384 + gc] + xr[384 + gc]);
        cst[r] = gf * cst[r] + gi * gg;
        const float hv = go * tanhf(cst[r]);
        hsum[r] += hv;
        hbuf[r][gc] = (_Float16)hv;
        if (mode == 1) hs_out[((size_t)r * 512 + t) * 128 + gc] = (_Float16)hv;
      }
    }
    __syncthreads();
  }

  if (mode == 0 && active) {
#pragma unroll
    for (int r = 0; r < 8; ++r) enc_mean[r * 128 + gc] = hsum[r] * (1.f / 512.f);
  }
}

// ---------------------------------------------------------------------------
// Small glue kernels
// ---------------------------------------------------------------------------
__global__ void gather_emb_f16(const int* __restrict__ ids,
                               const float* __restrict__ emb,
                               _Float16* __restrict__ out, int rows) {
  const int idx = blockIdx.x * blockDim.x + threadIdx.x;
  if (idx >= rows * 128) return;
  const int r = idx >> 7, c = idx & 127;
  out[idx] = (_Float16)emb[(size_t)ids[r] * 128 + c];
}

__global__ void f32_to_f16(const float* __restrict__ in,
                           _Float16* __restrict__ out, int n) {
  const int idx = blockIdx.x * blockDim.x + threadIdx.x;
  if (idx < n) out[idx] = (_Float16)in[idx];
}

__global__ void deg_count(const int* __restrict__ dst, float* __restrict__ deg,
                          int E) {
  const int e = blockIdx.x * blockDim.x + threadIdx.x;
  if (e < E) atomicAdd(&deg[dst[e]], 1.0f);
}

__global__ void make_dinv(float* __restrict__ deg, int n) {
  const int i = blockIdx.x * blockDim.x + threadIdx.x;
  if (i < n) deg[i] = rsqrtf(deg[i] + 1.0f);  // +1 for self-loop
}

// agg[dst] += h[src] * dinv[src]*dinv[dst] ; 1 thread per (edge, 4 columns)
__global__ void gcn_scatter(const int* __restrict__ src,
                            const int* __restrict__ dst,
                            const float* __restrict__ dinv,
                            const float* __restrict__ h,
                            float* __restrict__ agg, int E) {
  const int idx = blockIdx.x * blockDim.x + threadIdx.x;
  const int e = idx >> 5;
  if (e >= E) return;
  const int c4 = (idx & 31) * 4;
  const int s = src[e], d = dst[e];
  const float nm = dinv[s] * dinv[d];
  const v4f hv = *(const v4f*)(h + (size_t)s * 128 + c4);
  float* ap = agg + (size_t)d * 128 + c4;
  atomicAdd(ap + 0, hv.x * nm);
  atomicAdd(ap + 1, hv.y * nm);
  atomicAdd(ap + 2, hv.z * nm);
  atomicAdd(ap + 3, hv.w * nm);
}

// x = agg + h*dinv^2 + b ; also emit f16 copy for the next WMMA GEMM
__global__ void gcn_combine(const float* __restrict__ agg,
                            const float* __restrict__ h,
                            const float* __restrict__ dinv,
                            const float* __restrict__ b,
                            float* __restrict__ xout,
                            _Float16* __restrict__ xf16, int nn) {
  const int idx = blockIdx.x * blockDim.x + threadIdx.x;
  if (idx >= nn * 128) return;
  const int row = idx >> 7, c = idx & 127;
  const float di = dinv[row];
  const float v = agg[idx] + h[idx] * di * di + b[c];
  xout[idx] = v;
  xf16[idx] = (_Float16)v;
}

// deterministic column mean over N nodes: 1 block per column
__global__ __launch_bounds__(256) void col_mean(const float* __restrict__ x,
                                                float* __restrict__ out,
                                                int nn) {
  __shared__ float red[256];
  const int c = blockIdx.x;
  float s = 0.f;
  for (int i = threadIdx.x; i < nn; i += 256) s += x[(size_t)i * 128 + c];
  red[threadIdx.x] = s;
  __syncthreads();
  for (int stride = 128; stride > 0; stride >>= 1) {
    if (threadIdx.x < stride) red[threadIdx.x] += red[threadIdx.x + stride];
    __syncthreads();
  }
  if (threadIdx.x == 0) out[c] = red[0] / (float)nn;
}

// fused[b,j] = [enc_mean | graph_mean] . Wf[j,:] + bf[j]   (8x128, K=256)
__global__ void fuse_proj(const float* __restrict__ encm,
                          const float* __restrict__ gmean,
                          const float* __restrict__ Wf,
                          const float* __restrict__ bfv,
                          float* __restrict__ fused) {
  const int idx = blockIdx.x * blockDim.x + threadIdx.x;
  if (idx >= 8 * 128) return;
  const int b = idx >> 7, j = idx & 127;
  const float* w = Wf + (size_t)j * 256;
  float s = bfv[j];
  for (int k = 0; k < 128; ++k) s += encm[b * 128 + k] * w[k];
  for (int k = 0; k < 128; ++k) s += gmean[k] * w[128 + k];
  fused[idx] = s;
}

// decoder input gates (time-constant): xg_dec[b,n] = fused[b].W_ih_dec[n] + b_dec[n]
__global__ void dec_ingate(const float* __restrict__ fused,
                           const float* __restrict__ Wih,
                           const float* __restrict__ bd,
                           float* __restrict__ xgd) {
  const int idx = blockIdx.x * blockDim.x + threadIdx.x;
  if (idx >= 8 * 512) return;
  const int b = idx >> 9, n = idx & 511;
  const float* w = Wih + (size_t)n * 128;
  const float* f = fused + b * 128;
  float s = bd[n];
  for (int k = 0; k < 128; ++k) s += f[k] * w[k];
  xgd[(size_t)b * 512 + n] = s;
}

// ---------------------------------------------------------------------------
extern "C" void kernel_launch(void* const* d_in, const int* in_sizes, int n_in,
                              void* d_out, int out_size, void* d_ws,
                              size_t ws_size, hipStream_t stream) {
  const int*   json   = (const int*)  d_in[0];
  const int*   nodeid = (const int*)  d_in[1];
  const int*   ei     = (const int*)  d_in[2];
  const float* emb    = (const float*)d_in[3];
  const float* Wih_e  = (const float*)d_in[4];
  const float* Whh_e  = (const float*)d_in[5];
  const float* b_e    = (const float*)d_in[6];
  const float* W1     = (const float*)d_in[7];
  const float* b1     = (const float*)d_in[8];
  const float* W2     = (const float*)d_in[9];
  const float* b2     = (const float*)d_in[10];
  const float* Wf     = (const float*)d_in[11];
  const float* bfv    = (const float*)d_in[12];
  const float* Wih_d  = (const float*)d_in[13];
  const float* Whh_d  = (const float*)d_in[14];
  const float* b_d    = (const float*)d_in[15];
  const float* Wo     = (const float*)d_in[16];
  const float* bo     = (const float*)d_in[17];

  const int E  = in_sizes[2] / 2;        // 600000
  const int NN = in_sizes[1];            // 50000 nodes
  const int MS = in_sizes[0];            // B*S = 4096 sequence rows
  const int V  = 32000;
  const int* esrc = ei;
  const int* edst = ei + E;

  // ---- workspace layout ----
  char* base = (char*)d_ws;
  size_t off = 0;
  auto alloc = [&](size_t bytes) -> void* {
    void* r = base + off;
    off = (off + bytes + 255) & ~(size_t)255;
    return r;
  };
  _Float16* jsonEmb16 = (_Float16*)alloc((size_t)MS * 128 * 2);
  _Float16* nodeX16   = (_Float16*)alloc((size_t)NN * 128 * 2);  // reused for x1/x2 f16
  float*    xg_enc    = (float*)   alloc((size_t)MS * 512 * 4);
  float*    enc_mean  = (float*)   alloc(8 * 128 * 4);
  float*    h_gcn     = (float*)   alloc((size_t)NN * 128 * 4);
  float*    agg       = (float*)   alloc((size_t)NN * 128 * 4);
  float*    x_gcn     = (float*)   alloc((size_t)NN * 128 * 4);
  float*    deg       = (float*)   alloc((size_t)NN * 4);        // becomes dinv
  float*    gmean     = (float*)   alloc(128 * 4);
  float*    fused     = (float*)   alloc(8 * 128 * 4);
  float*    xg_dec    = (float*)   alloc(8 * 512 * 4);
  _Float16* hs_dec16  = (_Float16*)alloc((size_t)MS * 128 * 2);
  _Float16* Wih_e16   = (_Float16*)alloc((size_t)512 * 128 * 2);
  _Float16* W1_16     = (_Float16*)alloc((size_t)128 * 128 * 2);
  _Float16* W2_16     = (_Float16*)alloc((size_t)128 * 128 * 2);
  _Float16* Wo_16     = (_Float16*)alloc((size_t)V * 128 * 2);
  (void)ws_size; (void)n_in; (void)out_size;

  const int T = 256;
  auto blocks = [](long long n, int t) { return (unsigned)((n + t - 1) / t); };

  // ---- weight conversions + embedding gathers ----
  f32_to_f16<<<blocks(512 * 128, T), T, 0, stream>>>(Wih_e, Wih_e16, 512 * 128);
  f32_to_f16<<<blocks(128 * 128, T), T, 0, stream>>>(W1, W1_16, 128 * 128);
  f32_to_f16<<<blocks(128 * 128, T), T, 0, stream>>>(W2, W2_16, 128 * 128);
  f32_to_f16<<<blocks((long long)V * 128, T), T, 0, stream>>>(Wo, Wo_16, V * 128);
  gather_emb_f16<<<blocks((long long)MS * 128, T), T, 0, stream>>>(json, emb, jsonEmb16, MS);
  gather_emb_f16<<<blocks((long long)NN * 128, T), T, 0, stream>>>(nodeid, emb, nodeX16, NN);

  // ---- encoder: xg = emb @ W_ih^T + b (async-LDS WMMA), then recurrence ----
  gemm_nt_k128_lds<false><<<dim3(blocks(512, 256), blocks(MS, 32)), T, 0, stream>>>(
      jsonEmb16, Wih_e16, b_e, xg_enc, MS, 512);
  lstm_seq<<<1, T, 0, stream>>>(xg_enc, Whh_e, enc_mean, nullptr, 0);

  // ---- GCN layer 1 ----
  hipMemsetAsync(deg, 0, (size_t)NN * 4, stream);
  hipMemsetAsync(agg, 0, (size_t)NN * 128 * 4, stream);
  gemm_nt_k128<<<dim3(1, blocks(NN, 64)), T, 0, stream>>>(
      nodeX16, W1_16, nullptr, h_gcn, NN, 128);
  deg_count<<<blocks(E, T), T, 0, stream>>>(edst, deg, E);
  make_dinv<<<blocks(NN, T), T, 0, stream>>>(deg, NN);
  gcn_scatter<<<blocks((long long)E * 32, T), T, 0, stream>>>(
      esrc, edst, deg, h_gcn, agg, E);
  gcn_combine<<<blocks((long long)NN * 128, T), T, 0, stream>>>(
      agg, h_gcn, deg, b1, x_gcn, nodeX16, NN);

  // ---- GCN layer 2 ----
  hipMemsetAsync(agg, 0, (size_t)NN * 128 * 4, stream);
  gemm_nt_k128<<<dim3(1, blocks(NN, 64)), T, 0, stream>>>(
      nodeX16, W2_16, nullptr, h_gcn, NN, 128);
  gcn_scatter<<<blocks((long long)E * 32, T), T, 0, stream>>>(
      esrc, edst, deg, h_gcn, agg, E);
  gcn_combine<<<blocks((long long)NN * 128, T), T, 0, stream>>>(
      agg, h_gcn, deg, b2, x_gcn, nodeX16, NN);
  col_mean<<<128, T, 0, stream>>>(x_gcn, gmean, NN);

  // ---- fusion + decoder ----
  fuse_proj<<<4, T, 0, stream>>>(enc_mean, gmean, Wf, bfv, fused);
  dec_ingate<<<16, T, 0, stream>>>(fused, Wih_d, b_d, xg_dec);
  lstm_seq<<<1, T, 0, stream>>>(xg_dec, Whh_d, nullptr, hs_dec16, 1);

  // ---- output projection: [4096,128] @ [128,32000] + bo -> d_out (NT) ----
  gemm_nt_k128_lds<true><<<dim3(blocks(V, 256), blocks(MS, 32)), T, 0, stream>>>(
      hs_dec16, Wo_16, bo, (float*)d_out, MS, V);
}
